// FullyVectorizedSimilaritySmoothing_34248069218791
// MI455X (gfx1250) — matrix operand order repaired and checked by reading, not compiled, not run on hardware
//
#include <hip/hip_runtime.h>
#include <hip/hip_bf16.h>

#define HDIM 768
#define PDIM 256
#define BATCH 8
#define LSEQ 2048

typedef __attribute__((ext_vector_type(16))) __bf16 v16bf;
typedef __attribute__((ext_vector_type(8)))  __bf16 v8bf;
typedef __attribute__((ext_vector_type(8)))  float  v8f;

static __device__ __forceinline__ v8f wmma_bf16(v16bf a, v16bf b, v8f c) {
  // D = A(16x32 bf16) * B(32x16 bf16) + C(16x16 f32)
  return __builtin_amdgcn_wmma_f32_16x16x32_bf16(false, a, false, b, (short)0, c,
                                                 false, false);
}

static __device__ __forceinline__ v16bf load_bf16x16_pair(const __bf16* p0, const __bf16* p1) {
  v8bf lo = *(const v8bf*)p0;
  v8bf hi = *(const v8bf*)p1;
  v16bf r;
#pragma unroll
  for (int i = 0; i < 8; ++i) { r[i] = lo[i]; r[8 + i] = hi[i]; }
  return r;
}

// Async global -> LDS copy, 16B per lane per instruction, tracked by ASYNCcnt.
// LDS address = low 32 bits of the generic pointer (flat->LDS uses addr[31:0]).
static __device__ __forceinline__ void async_ld_b128(void* lds, const void* gsrc) {
  const unsigned int laddr = (unsigned int)(unsigned long long)lds;
  asm volatile("global_load_async_to_lds_b128 %0, %1, off"
               :: "v"(laddr), "v"(gsrc) : "memory");
}
static __device__ __forceinline__ void wait_async_24() {
  asm volatile("s_wait_asynccnt 24" ::: "memory");
}
static __device__ __forceinline__ void wait_async_0() {
  asm volatile("s_wait_asynccnt 0" ::: "memory");
}

// ---------------------------------------------------------------------------
// Y[M,H] = bf16( X[M,H] @ W^T[H,H] ),  M = B*L.  One block = 64x64 out tile,
// 4 waves, each wave does a 16x64 strip with 4 C fragments.
// ---------------------------------------------------------------------------
__global__ __launch_bounds__(128) void proj_gemm_bf16(const float* __restrict__ X,
                                                      const float* __restrict__ W,
                                                      __bf16* __restrict__ Y) {
  const int lane = threadIdx.x & 31;
  const int wave = threadIdx.x >> 5;
  const int hf   = lane >> 4;
  const int m16  = lane & 15;
  const int rowBase = blockIdx.x * 64 + wave * 16;
  const int colBase = blockIdx.y * 64;

  v8f c[4];
#pragma unroll
  for (int nt = 0; nt < 4; ++nt)
#pragma unroll
    for (int j = 0; j < 8; ++j) c[nt][j] = 0.f;

  const int row = rowBase + m16;
  for (int k0 = 0; k0 < HDIM; k0 += 32) {
    // A fragment (contiguous K): a[0..15] -> K=k0+hf*16..
    v16bf a;
    {
      const float* pa = X + (size_t)row * HDIM + k0 + hf * 16;
#pragma unroll
      for (int i = 0; i < 8; ++i) a[i] = (__bf16)pa[i];
#pragma unroll
      for (int i = 0; i < 8; ++i) a[8 + i] = (__bf16)pa[8 + i];
    }
#pragma unroll
    for (int nt = 0; nt < 4; ++nt) {
      // B fragment: lane = output column; b[i] = W^T[k0+hf*16+i][col] = W[col][...]
      const int col = colBase + nt * 16 + m16;
      const float* pb = W + (size_t)col * HDIM + k0 + hf * 16;
      v16bf b;
#pragma unroll
      for (int i = 0; i < 16; ++i) b[i] = (__bf16)pb[i];
      c[nt] = wmma_bf16(a, b, c[nt]);
    }
  }
  // C layout: VGPR j holds row (j + hf*8), column m16
#pragma unroll
  for (int nt = 0; nt < 4; ++nt)
#pragma unroll
    for (int j = 0; j < 8; ++j) {
      const int r  = rowBase + hf * 8 + j;
      const int cc = colBase + nt * 16 + m16;
      Y[(size_t)r * HDIM + cc] = (__bf16)c[nt][j];
    }
}

// ---------------------------------------------------------------------------
// temps[row] = relu( X[row,:] . Wt ) + 0.01 ; one wave per row
// ---------------------------------------------------------------------------
__global__ __launch_bounds__(128) void temps_kernel(const float* __restrict__ X,
                                                    const float* __restrict__ Wt,
                                                    float* __restrict__ temps) {
  const int lane = threadIdx.x & 31;
  const int wave = threadIdx.x >> 5;
  const int row  = blockIdx.x * 4 + wave;
  const float* px = X + (size_t)row * HDIM;
  float s = 0.f;
  for (int t = lane; t < HDIM; t += 32) s += px[t] * Wt[t];
#pragma unroll
  for (int d = 16; d >= 1; d >>= 1) s += __shfl_xor(s, d, 32);
  if (lane == 0) temps[row] = fmaxf(s, 0.f) + 0.01f;
}

__global__ void cvt_bf16_kernel(const float* __restrict__ src, __bf16* __restrict__ dst, int n) {
  int i = blockIdx.x * blockDim.x + threadIdx.x;
  if (i < n) dst[i] = (__bf16)src[i];
}

// ---------------------------------------------------------------------------
// Flash-attention with question-type masking + per-row temperature.
// Block = 4 waves = 64 query rows of one batch; wave owns 16 rows and the
// full P=256 accumulator (16 C fragments). K tiles are double-buffered via
// global_load_async_to_lds_b128 (ASYNCcnt), prefetching tile kc+32 while the
// 48 WMMAs of tile kc run.
// ---------------------------------------------------------------------------
#define SM_QS 0
#define SM_K0 (64 * HDIM * 2)                 //  98304
#define SM_K1 (SM_K0 + 32 * HDIM * 2)         // 147456
#define SM_VT (SM_K1 + 32 * HDIM * 2)         // 196608
#define SM_PS (SM_VT + PDIM * 32 * 2)         // 212992
#define SM_TOTAL (SM_PS + 4 * 16 * 32 * 2)    // 217088

// 32x768 bf16 tile = 49152 B = 3072 x 16B; 128 threads -> 24 async ops/thread
static __device__ __forceinline__ void stage_k_async(__bf16* dstLds, const __bf16* src,
                                                     int tid) {
  char* l = (char*)dstLds;
  const char* g = (const char*)src;
#pragma unroll 4
  for (int i = tid; i < 3072; i += 128) async_ld_b128(l + i * 16, g + i * 16);
}

__global__ __launch_bounds__(128, 1) void attn_kernel(
    const __bf16* __restrict__ Qbf, const __bf16* __restrict__ Kbf,
    const __bf16* __restrict__ Vbf, const float* __restrict__ temps,
    const int* __restrict__ questions, const int* __restrict__ mask,
    const float* __restrict__ param, float* __restrict__ outS) {
  extern __shared__ char smem[];
  __bf16* Qs  = (__bf16*)(smem + SM_QS);  // 64 x 768
  __bf16* Ks0 = (__bf16*)(smem + SM_K0);  // 32 x 768 (buffer A)
  __bf16* Ks1 = (__bf16*)(smem + SM_K1);  // 32 x 768 (buffer B)
  __bf16* Vt  = (__bf16*)(smem + SM_VT);  // 256 x 32 (transposed V tile)
  __bf16* Ps  = (__bf16*)(smem + SM_PS);  // per-wave 16 x 32 probs

  const int tid  = threadIdx.x;
  const int lane = tid & 31;
  const int wave = tid >> 5;
  const int hf   = lane >> 4;
  const int n16  = lane & 15;
  const int b    = blockIdx.y;
  const int qbase = blockIdx.x * 64;
  const __bf16* Ksrc = Kbf + (size_t)(b * LSEQ) * HDIM;

  // prologue: async prefetch of K tile 0
  stage_k_async(Ks0, Ksrc, tid);

  // stage Q tile (contiguous 64*768 bf16)
  {
    const uint4* src = (const uint4*)(Qbf + (size_t)(b * LSEQ + qbase) * HDIM);
    uint4* dst = (uint4*)Qs;
    for (int i = tid; i < (64 * HDIM) / 8; i += 128) dst[i] = src[i];
  }

  // per-row temperature + question id (rows j + hf*8 of this wave's tile)
  float tq[8]; int qq[8];
  const int qrow0 = qbase + wave * 16 + hf * 8;
#pragma unroll
  for (int j = 0; j < 8; ++j) {
    const int r = b * LSEQ + qrow0 + j;
    tq[j] = temps[r];
    qq[j] = questions[r];
  }

  float mrow[8], lrow[8];
#pragma unroll
  for (int j = 0; j < 8; ++j) { mrow[j] = -1e30f; lrow[j] = 0.f; }
  v8f o[16];
#pragma unroll
  for (int nt = 0; nt < 16; ++nt)
#pragma unroll
    for (int j = 0; j < 8; ++j) o[nt][j] = 0.f;

  const __bf16* Qrow = Qs + (wave * 16 + n16) * HDIM;

  for (int kc = 0; kc < LSEQ; kc += 32) {
    const int cur = (kc >> 5) & 1;
    __bf16* Kcur = cur ? Ks1 : Ks0;
    __bf16* Knxt = cur ? Ks0 : Ks1;

    __syncthreads();  // prior iteration finished reading Knxt / Vt / Ps

    {   // V tile, transposed into [p][key] (scalar path: needs transpose)
      const __bf16* vsrc = Vbf + (size_t)(b * LSEQ + kc) * PDIM;
      for (int e = tid; e < (32 * PDIM) / 8; e += 128) {
        const int key = e >> 5;
        const int pv  = (e & 31) * 8;
        v8bf v = *(const v8bf*)(vsrc + key * PDIM + pv);
#pragma unroll
        for (int i = 0; i < 8; ++i) Vt[(pv + i) * 32 + key] = v[i];
      }
    }

    // prefetch next K tile, then wait for the CURRENT tile (in-order: allowing
    // 24 outstanding per wave leaves exactly the next tile in flight)
    if (kc + 32 < LSEQ) {
      stage_k_async(Knxt, Ksrc + (size_t)(kc + 32) * HDIM, tid);
      wait_async_24();
    } else {
      wait_async_0();
    }
    __syncthreads();  // K tile + Vt visible to all waves

    // S = Q @ K^T  (two 16x16 fragments: keys kc..+15 and kc+16..+31)
    v8f s0, s1;
#pragma unroll
    for (int j = 0; j < 8; ++j) { s0[j] = 0.f; s1[j] = 0.f; }
    for (int h0 = 0; h0 < HDIM; h0 += 32) {
      v16bf a = load_bf16x16_pair(Qrow + h0 + hf * 16, Qrow + h0 + hf * 16 + 8);
      v16bf b0 = *(const v16bf*)(Kcur + n16 * HDIM + h0 + hf * 16);
      v16bf b1 = *(const v16bf*)(Kcur + (n16 + 16) * HDIM + h0 + hf * 16);
      s0 = wmma_bf16(a, b0, s0);
      s1 = wmma_bf16(a, b1, s1);
    }

    const int qk0 = questions[b * LSEQ + kc + n16];
    const int qk1 = questions[b * LSEQ + kc + 16 + n16];

    float p0[8], p1[8], alpha[8];
#pragma unroll
    for (int j = 0; j < 8; ++j) {
      const bool ok0 = (qk0 == qq[j]);
      const bool ok1 = (qk1 == qq[j]);
      const float inv_t = 1.f / tq[j];
      float v0 = ok0 ? s0[j] * inv_t : -1e30f;
      float v1 = ok1 ? s1[j] * inv_t : -1e30f;
      float mx = fmaxf(v0, v1);
#pragma unroll
      for (int d = 1; d < 16; d <<= 1) mx = fmaxf(mx, __shfl_xor(mx, d, 32));
      const float mn = fmaxf(mrow[j], mx);
      const float al = __expf(mrow[j] - mn);
      const float e0 = ok0 ? __expf(v0 - mn) : 0.f;
      const float e1 = ok1 ? __expf(v1 - mn) : 0.f;
      float sm = e0 + e1;
#pragma unroll
      for (int d = 1; d < 16; d <<= 1) sm += __shfl_xor(sm, d, 32);
      lrow[j] = lrow[j] * al + sm;
      mrow[j] = mn;
      alpha[j] = al;
      p0[j] = e0; p1[j] = e1;
    }

#pragma unroll
    for (int nt = 0; nt < 16; ++nt)
#pragma unroll
      for (int j = 0; j < 8; ++j) o[nt][j] *= alpha[j];

    // stage probabilities (C layout -> A layout via per-wave LDS tile)
    __bf16* prow = Ps + wave * 16 * 32;
#pragma unroll
    for (int j = 0; j < 8; ++j) {
      prow[(hf * 8 + j) * 32 + n16]      = (__bf16)p0[j];
      prow[(hf * 8 + j) * 32 + 16 + n16] = (__bf16)p1[j];
    }
    __syncthreads();

    // O += P(16x32) @ V(32x256): 16 WMMA accumulations
    const __bf16* pr = Ps + wave * 16 * 32 + n16 * 32;
    v16bf pa = load_bf16x16_pair(pr + hf * 16, pr + hf * 16 + 8);
#pragma unroll
    for (int nt = 0; nt < 16; ++nt) {
      v16bf bv = *(const v16bf*)(Vt + (nt * 16 + n16) * 32 + hf * 16);
      o[nt] = wmma_bf16(pa, bv, o[nt]);
    }
  }

  // epilogue: normalize, apply smoothing mask (identity rows where mask==0)
#pragma unroll
  for (int j = 0; j < 8; ++j) {
    const int r = qbase + wave * 16 + hf * 8 + j;
    const int msk = mask[b * LSEQ + r];
    const float inv = 1.f / lrow[j];
#pragma unroll
    for (int nt = 0; nt < 16; ++nt) {
      const int col = nt * 16 + n16;
      float val = o[nt][j] * inv;
      if (!msk) val = param[((size_t)(b * LSEQ) + r) * PDIM + col];
      outS[((size_t)(b * LSEQ) + r) * PDIM + col] = val;
    }
  }
}

// ---------------------------------------------------------------------------
extern "C" void kernel_launch(void* const* d_in, const int* in_sizes, int n_in,
                              void* d_out, int out_size, void* d_ws, size_t ws_size,
                              hipStream_t stream) {
  (void)in_sizes; (void)n_in; (void)out_size; (void)ws_size;
  const float* hs        = (const float*)d_in[0];
  const float* ps        = (const float*)d_in[1];
  const int*   questions = (const int*)d_in[2];
  const int*   mask      = (const int*)d_in[3];
  const float* Wq        = (const float*)d_in[4];
  const float* Wk        = (const float*)d_in[5];
  const float* Wt        = (const float*)d_in[6];
  float* out = (float*)d_out;

  char* ws = (char*)d_ws;
  const size_t qk_bytes = (size_t)BATCH * LSEQ * HDIM * 2;
  const size_t v_bytes  = (size_t)BATCH * LSEQ * PDIM * 2;
  __bf16* Qbf   = (__bf16*)ws;
  __bf16* Kbf   = (__bf16*)(ws + qk_bytes);
  __bf16* Vbf   = (__bf16*)(ws + 2 * qk_bytes);
  float*  temps = (float*)(ws + 2 * qk_bytes + v_bytes);

  // output[0]: hidden_states passthrough
  hipMemcpyAsync(out, hs, (size_t)BATCH * LSEQ * HDIM * sizeof(float),
                 hipMemcpyDeviceToDevice, stream);

  // Q / K projections (bf16 WMMA GEMMs)
  dim3 gp((BATCH * LSEQ) / 64, HDIM / 64);
  proj_gemm_bf16<<<gp, 128, 0, stream>>>(hs, Wq, Qbf);
  proj_gemm_bf16<<<gp, 128, 0, stream>>>(hs, Wk, Kbf);

  // temperatures
  temps_kernel<<<(BATCH * LSEQ) / 4, 128, 0, stream>>>(hs, Wt, temps);

  // V -> bf16
  const int nv = BATCH * LSEQ * PDIM;
  cvt_bf16_kernel<<<(nv + 255) / 256, 256, 0, stream>>>(ps, Vbf, nv);

  // masked flash attention + smoothing
  dim3 ga(LSEQ / 64, BATCH);
  attn_kernel<<<ga, 128, SM_TOTAL, stream>>>(Qbf, Kbf, Vbf, temps, questions, mask,
                                             ps, out + (size_t)BATCH * LSEQ * HDIM);
}
